// qkv_module_82025285419092
// MI455X (gfx1250) — 8-bit matrix operand order repaired and checked by reading, compiled, not run on hardware
//
#include <hip/hip_runtime.h>
#include <stdint.h>

typedef __attribute__((ext_vector_type(16))) int   v16i;
typedef __attribute__((ext_vector_type(8)))  float v8f;
typedef __attribute__((ext_vector_type(4)))  float v4f;

#define DIN   1024
#define DOUT  3072
#define MROWS 16384

#define FP8LUT_LO 0x4E4C4A4844403800ull
#define FP8LUT_HI 0x5756555453525150ull

__device__ __forceinline__ uint32_t q_to_fp8(int q) {
    uint64_t lut = (q < 8) ? FP8LUT_LO : FP8LUT_HI;
    return (uint32_t)((lut >> ((q & 7) * 8)) & 0xFF);
}

__global__ __launch_bounds__(256)
void quant_x_kernel(const float* __restrict__ x,
                    const float* __restrict__ in_lb,
                    const float* __restrict__ in_ub,
                    uint8_t* __restrict__ xq,
                    float* __restrict__ Sx) {
    const int row = blockIdx.x;
    const int tid = threadIdx.x;
    const float lb = in_lb[0];
    const float ub = in_ub[0];
    const float delta = (ub - lb) * (1.0f / 15.0f);
    const float inv = (delta != 0.0f) ? (1.0f / delta) : 0.0f;

    const v4f xv =
        __builtin_nontemporal_load((const v4f*)(x + (size_t)row * DIN + tid * 4));
    int s = 0;
    uint32_t packed = 0;
#pragma unroll
    for (int j = 0; j < 4; ++j) {
        float c = fminf(fmaxf(xv[j], lb), ub);
        int q = (int)__builtin_rintf((c - lb) * inv);
        s += q;
        packed |= q_to_fp8(q) << (8 * j);
    }
    *(uint32_t*)(xq + (size_t)row * DIN + tid * 4) = packed;

    __shared__ int red[256];
    red[tid] = s;
    __syncthreads();
#pragma unroll
    for (int off = 128; off > 0; off >>= 1) {
        if (tid < off) red[tid] += red[tid + off];
        __syncthreads();
    }
    if (tid == 0) Sx[row] = (float)red[0];
}

__global__ __launch_bounds__(256)
void quant_w_kernel(const float* __restrict__ w,
                    const float* __restrict__ w_lb,
                    const float* __restrict__ w_ub,
                    uint8_t* __restrict__ wq,
                    float* __restrict__ Sw) {
    const int row = blockIdx.x;
    const int tid = threadIdx.x;
    const float lb = w_lb[row];
    const float ub = w_ub[row];
    const float delta = (ub - lb) * (1.0f / 15.0f);
    const float inv = (delta != 0.0f) ? (1.0f / delta) : 0.0f;

    const v4f xv =
        __builtin_nontemporal_load((const v4f*)(w + (size_t)row * DIN + tid * 4));
    int s = 0;
    uint32_t packed = 0;
#pragma unroll
    for (int j = 0; j < 4; ++j) {
        float c = fminf(fmaxf(xv[j], lb), ub);
        int q = (int)__builtin_rintf((c - lb) * inv);
        s += q;
        packed |= q_to_fp8(q) << (8 * j);
    }
    *(uint32_t*)(wq + (size_t)row * DIN + tid * 4) = packed;

    __shared__ int red[256];
    red[tid] = s;
    __syncthreads();
#pragma unroll
    for (int off = 128; off > 0; off >>= 1) {
        if (tid < off) red[tid] += red[tid + off];
        __syncthreads();
    }
    if (tid == 0) Sw[row] = (float)red[0];
}

__global__ __launch_bounds__(256)
void wmma_fp8_gemm_kernel(const uint8_t* __restrict__ xq,
                          const uint8_t* __restrict__ wq,
                          const float* __restrict__ Sx,
                          const float* __restrict__ Sw,
                          const float* __restrict__ bias,
                          const float* __restrict__ in_lb,
                          const float* __restrict__ in_ub,
                          const float* __restrict__ w_lb,
                          const float* __restrict__ w_ub,
                          float* __restrict__ out) {
    const int lane = threadIdx.x & 31;
    const int wave = threadIdx.x >> 5;
    const int wm = wave >> 2;
    const int wn = wave & 3;
    const int M0 = blockIdx.y * 64 + wm * 32;
    const int N0 = blockIdx.x * 256 + wn * 64;
    const int half = lane >> 4;
    const int l16 = lane & 15;

    v8f acc[2][4] = {};

    const uint8_t* aBase0 = xq + (size_t)(M0 + l16) * DIN + half * 16;
    const uint8_t* aBase1 = aBase0 + (size_t)16 * DIN;
    const uint8_t* bBase[4];
#pragma unroll
    for (int tn = 0; tn < 4; ++tn)
        bBase[tn] = wq + (size_t)(N0 + tn * 16 + l16) * DIN + half * 16;

#pragma unroll 2
    for (int kk = 0; kk < DIN; kk += 128) {
        v16i Bm[4];
#pragma unroll
        for (int tn = 0; tn < 4; ++tn) {
#pragma unroll
            for (int blk = 0; blk < 4; ++blk) {
                uint4 q0 = *(const uint4*)(bBase[tn] + kk + blk * 32);
                Bm[tn][4 * blk + 0] = (int)q0.x;
                Bm[tn][4 * blk + 1] = (int)q0.y;
                Bm[tn][4 * blk + 2] = (int)q0.z;
                Bm[tn][4 * blk + 3] = (int)q0.w;
            }
        }
        v16i A0, A1;
#pragma unroll
        for (int chunk = 0; chunk < 2; ++chunk) {
#pragma unroll
            for (int g = 0; g < 4; ++g) {
                const int off = kk + chunk * 64 + (g >> 1) * 32 + (g & 1) * 8;
                uint2 d0 = *(const uint2*)(aBase0 + off);
                uint2 d1 = *(const uint2*)(aBase1 + off);
                const int v = chunk * 8 + 2 * g;
                A0[v] = (int)d0.x; A0[v + 1] = (int)d0.y;
                A1[v] = (int)d1.x; A1[v + 1] = (int)d1.y;
            }
        }
        __builtin_amdgcn_sched_barrier(0);
#pragma unroll
        for (int tn = 0; tn < 4; ++tn) {
            acc[0][tn] = __builtin_amdgcn_wmma_f32_16x16x128_fp8_fp8(
                A0, Bm[tn], (short)0, acc[0][tn], false, false);
            acc[1][tn] = __builtin_amdgcn_wmma_f32_16x16x128_fp8_fp8(
                A1, Bm[tn], (short)0, acc[1][tn], false, false);
        }
    }

    const float lx  = in_lb[0];
    const float uxv = in_ub[0];
    const float dxs = (uxv - lx) * (1.0f / 15.0f);
#pragma unroll
    for (int tn = 0; tn < 4; ++tn) {
        const int n = N0 + tn * 16 + l16;
        const float wl = w_lb[n];
        const float wu = w_ub[n];
        const float dw = (wu - wl) * (1.0f / 15.0f);
        const float cN = lx * dw * Sw[n] + (float)DIN * lx * wl + bias[n];
        const float a1 = dxs * dw;
        const float a2 = dxs * wl;
#pragma unroll
        for (int tm = 0; tm < 2; ++tm) {
#pragma unroll
            for (int r = 0; r < 8; ++r) {
                const int m = M0 + tm * 16 + half * 8 + r;
                __builtin_nontemporal_store(a1 * acc[tm][tn][r] + a2 * Sx[m] + cN,
                                            out + (size_t)m * DOUT + n);
            }
        }
    }
}

extern "C" void kernel_launch(void* const* d_in, const int* in_sizes, int n_in,
                              void* d_out, int out_size, void* d_ws, size_t ws_size,
                              hipStream_t stream) {
    const float* x      = (const float*)d_in[0];
    const float* weight = (const float*)d_in[1];
    const float* bias   = (const float*)d_in[2];
    const float* in_lb  = (const float*)d_in[3];
    const float* in_ub  = (const float*)d_in[4];
    const float* w_lb   = (const float*)d_in[5];
    const float* w_ub   = (const float*)d_in[6];
    float* out = (float*)d_out;

    uint8_t* ws = (uint8_t*)d_ws;
    uint8_t* xq = ws;
    uint8_t* wq = ws + (size_t)MROWS * DIN;
    float*   Sx = (float*)(wq + (size_t)DOUT * DIN);
    float*   Sw = Sx + MROWS;

    quant_x_kernel<<<MROWS, 256, 0, stream>>>(x, in_lb, in_ub, xq, Sx);
    quant_w_kernel<<<DOUT, 256, 0, stream>>>(weight, w_lb, w_ub, wq, Sw);

    dim3 grid(DOUT / 256, MROWS / 64);
    wmma_fp8_gemm_kernel<<<grid, 256, 0, stream>>>(
        xq, wq, Sx, Sw, bias, in_lb, in_ub, w_lb, w_ub, out);
}
